// MultiHeadLatentAttention_69372311765402
// MI455X (gfx1250) — matrix operand order repaired and checked by reading, not compiled, not run on hardware
//
#include <hip/hip_runtime.h>
#include <hip/hip_bf16.h>
#include <cstdint>

typedef __attribute__((ext_vector_type(16))) _Float16 v16h;
typedef __attribute__((ext_vector_type(8)))  float    v8f;

#define H_HEADS 16
#define D_HD    128
#define D_HR    64
#define D_QK    192   /* D_HD + D_HR */
#define DMODEL  2048
#define DLAT    512

__device__ __forceinline__ v8f wmma_f16(v16h a, v16h b, v8f c) {
  // v_wmma_f32_16x16x32_f16: D = A(16x32 f16) * B(32x16 f16) + C(16x16 f32)
  return __builtin_amdgcn_wmma_f32_16x16x32_f16(false, a, false, b, (short)0, c,
                                                false, false);
}

enum OutMode {
  OM_F16 = 0,          // plain f16, row stride ldo
  OM_HEAD192_LO = 1,   // col c -> head c>>7, dim c&127, into (H,S,192) at offset 0
  OM_HEAD192_HI = 2,   // col c -> head c>>6, dim c&63,  into (H,S,192) at offset 128
  OM_VT = 3,           // col c -> head c>>7, dim c&127, into (H,128,S) transposed
  OM_F32 = 4           // plain f32, row stride ldo
};

// ---- WMMA operand loaders (wave32; this code's contiguous k order k = 16*(lane/16) + s, same for A and B) ----

// A-matrix 16x32 (MxK), 16-bit: lanes 0-15 row M=lane hold K=0..15,
// lanes 16-31 row M=lane-16 hold K=16..31.
template <typename TA>
__device__ __forceinline__ v16h load_a_tile(const TA* __restrict__ A, int lda,
                                            int row0, int k0, int lane) {
  const int r  = lane & 15;
  const int kb = (lane >> 4) * 16;
  const TA* p = A + (size_t)(row0 + r) * lda + k0 + kb;
  v16h a;
#pragma unroll
  for (int i = 0; i < 8; ++i) {
    a[i]     = (_Float16)p[i];
    a[8 + i] = (_Float16)p[8 + i];
  }
  return a;
}

// B-matrix 32x16 (KxN), 16-bit: lanes 0-15 col N=lane hold K=0..15,
// lanes 16-31 col N=lane-16 hold K=16..31. Source rows are K-contiguous
// (weights are (N,K) row-major), so each lane reads 16 contiguous values.
__device__ __forceinline__ v16h load_b_tile_f32(const float* __restrict__ B,
                                                int ldb, int n0, int k0, int lane) {
  const int n  = lane & 15;
  const int kb = (lane >> 4) * 16;
  const float* p = B + (size_t)(n0 + n) * ldb + k0 + kb;
  v16h b;
#pragma unroll
  for (int i = 0; i < 16; ++i) b[i] = (_Float16)p[i];
  return b;
}

// Contiguous 16-half B tile (f16 source)
__device__ __forceinline__ v16h load_b16(const _Float16* __restrict__ p) {
  v16h b;
#pragma unroll
  for (int i = 0; i < 16; ++i) b[i] = p[i];
  return b;
}

// ---- Generic WMMA GEMM: out = A (MxK) * B(N,K)^T, wave tile 32x64 ----
// K must be a multiple of 64 (holds for 2048 and 512). Double-buffered k-loop.

template <typename TA, int OMODE>
__global__ __launch_bounds__(128) void gemm_wmma_kernel(
    const TA* __restrict__ A, int lda, const float* __restrict__ B, int ldb,
    void* __restrict__ outv, int M, int N, int K, int ldo, int S) {
  const int lane = threadIdx.x & 31;
  const int wavesN = N >> 6;
  const int total = (M >> 5) * wavesN;
  const int w = blockIdx.x * 4 + (threadIdx.x >> 5);
  if (w >= total) return;                  // wave-uniform: EXEC stays all-ones
  const int m0 = (w / wavesN) * 32;
  const int n0 = (w % wavesN) * 64;

  const v8f zero = {0.f, 0.f, 0.f, 0.f, 0.f, 0.f, 0.f, 0.f};
  v8f acc[2][4];
#pragma unroll
  for (int i = 0; i < 2; ++i)
#pragma unroll
    for (int j = 0; j < 4; ++j) acc[i][j] = zero;

  // prologue: operand set "c" holds k=0
  v16h a0c = load_a_tile<TA>(A, lda, m0, 0, lane);
  v16h a1c = load_a_tile<TA>(A, lda, m0 + 16, 0, lane);
  v16h bc[4];
#pragma unroll
  for (int nt = 0; nt < 4; ++nt) bc[nt] = load_b_tile_f32(B, ldb, n0 + nt * 16, 0, lane);

  for (int k0 = 0; k0 < K; k0 += 64) {
    // issue loads for k0+32 while set "c" computes
    v16h a0n = load_a_tile<TA>(A, lda, m0, k0 + 32, lane);
    v16h a1n = load_a_tile<TA>(A, lda, m0 + 16, k0 + 32, lane);
    v16h bn[4];
#pragma unroll
    for (int nt = 0; nt < 4; ++nt)
      bn[nt] = load_b_tile_f32(B, ldb, n0 + nt * 16, k0 + 32, lane);

#pragma unroll
    for (int nt = 0; nt < 4; ++nt) {
      acc[0][nt] = wmma_f16(a0c, bc[nt], acc[0][nt]);
      acc[1][nt] = wmma_f16(a1c, bc[nt], acc[1][nt]);
    }

    // issue loads for k0+64 while set "n" computes (uniform guard)
    if (k0 + 64 < K) {
      a0c = load_a_tile<TA>(A, lda, m0, k0 + 64, lane);
      a1c = load_a_tile<TA>(A, lda, m0 + 16, k0 + 64, lane);
#pragma unroll
      for (int nt = 0; nt < 4; ++nt)
        bc[nt] = load_b_tile_f32(B, ldb, n0 + nt * 16, k0 + 64, lane);
    }

#pragma unroll
    for (int nt = 0; nt < 4; ++nt) {
      acc[0][nt] = wmma_f16(a0n, bn[nt], acc[0][nt]);
      acc[1][nt] = wmma_f16(a1n, bn[nt], acc[1][nt]);
    }
  }

  // C/D layout: VGPR j -> row mb+j (mb = 0 for lanes 0-15, 8 for 16-31), col = lane&15
  const int nl = lane & 15;
  const int mb = (lane >> 4) * 8;
#pragma unroll
  for (int mt = 0; mt < 2; ++mt) {
#pragma unroll
    for (int nt = 0; nt < 4; ++nt) {
#pragma unroll
      for (int j = 0; j < 8; ++j) {
        const int r = m0 + mt * 16 + mb + j;
        const int c = n0 + nt * 16 + nl;
        const float val = acc[mt][nt][j];
        if (OMODE == OM_F16) {
          ((_Float16*)outv)[(size_t)r * ldo + c] = (_Float16)val;
        } else if (OMODE == OM_HEAD192_LO) {
          const int h = c >> 7, d = c & 127;
          ((_Float16*)outv)[((size_t)h * S + r) * D_QK + d] = (_Float16)val;
        } else if (OMODE == OM_HEAD192_HI) {
          const int h = c >> 6, d = c & 63;
          ((_Float16*)outv)[((size_t)h * S + r) * D_QK + D_HD + d] = (_Float16)val;
        } else if (OMODE == OM_VT) {
          const int h = c >> 7, d = c & 127;
          ((_Float16*)outv)[(size_t)(h * D_HD + d) * S + r] = (_Float16)val;
        } else {
          ((float*)outv)[(size_t)r * ldo + c] = val;
        }
      }
    }
  }
}

// ---- RoPE kernels ----

__global__ void rope_q_kernel(_Float16* __restrict__ qh, int S) {
  const int idx = blockIdx.x * blockDim.x + threadIdx.x;
  if (idx >= H_HEADS * S * (D_HR / 2)) return;
  const int j = idx & 31;
  const int s = (idx >> 5) % S;
  const int h = idx / (S * 32);
  const float inv_freq = __expf(-9.210340371976184f * (float)(2 * j) / 64.f);
  const float ang = (float)s * inv_freq;
  const float cs = cosf(ang), sn = sinf(ang);
  const size_t base = ((size_t)h * S + s) * D_QK + D_HD + 2 * j;
  const float x1 = (float)qh[base];
  const float x2 = (float)qh[base + 1];
  qh[base]     = (_Float16)(x1 * cs - x2 * sn);
  qh[base + 1] = (_Float16)(x1 * sn + x2 * cs);
}

__global__ void rope_kr_bcast_kernel(const _Float16* __restrict__ kRraw,
                                     _Float16* __restrict__ kh, int S) {
  const int idx = blockIdx.x * blockDim.x + threadIdx.x;
  if (idx >= S * (D_HR / 2)) return;
  const int j = idx & 31;
  const int s = idx >> 5;
  const float inv_freq = __expf(-9.210340371976184f * (float)(2 * j) / 64.f);
  const float ang = (float)s * inv_freq;
  const float cs = cosf(ang), sn = sinf(ang);
  const float x1 = (float)kRraw[(size_t)s * D_HR + 2 * j];
  const float x2 = (float)kRraw[(size_t)s * D_HR + 2 * j + 1];
  const _Float16 o1 = (_Float16)(x1 * cs - x2 * sn);
  const _Float16 o2 = (_Float16)(x1 * sn + x2 * cs);
#pragma unroll
  for (int h = 0; h < H_HEADS; ++h) {
    const size_t base = ((size_t)h * S + s) * D_QK + D_HD + 2 * j;
    kh[base]     = o1;
    kh[base + 1] = o2;
  }
}

// ---- Flash attention: one wave per (head, 16-row q tile) ----

__global__ __launch_bounds__(128) void mla_flash_attn_kernel(
    const _Float16* __restrict__ qh, const _Float16* __restrict__ kh,
    const _Float16* __restrict__ vT, _Float16* __restrict__ attnh, int S) {
  __shared__ _Float16 Pst[4][16 * 32];  // per-wave P-tile relayout buffer

  const int lane = threadIdx.x & 31;
  const int wv = threadIdx.x >> 5;
  const int qtiles = S >> 4;
  const int w = blockIdx.x * 4 + wv;
  if (w >= H_HEADS * qtiles) return;     // wave-uniform
  const int h = w / qtiles;
  const int m0 = (w % qtiles) * 16;

  const int nl = lane & 15;
  const int kb8 = (lane >> 4) * 8;
  const int kb16 = (lane >> 4) * 16;
  const int mb = kb8;

  // Q tile (16x192) as six A-layout fragments, resident for the whole pass
  v16h qa[6];
  {
    const _Float16* qrow = qh + ((size_t)h * S + m0 + nl) * D_QK;
#pragma unroll
    for (int t = 0; t < 6; ++t) {
#pragma unroll
      for (int i = 0; i < 8; ++i) {
        qa[t][i]     = qrow[t * 32 + kb16 + i];
        qa[t][8 + i] = qrow[t * 32 + kb16 + 8 + i];
      }
    }
  }

  const v8f zero = {0.f, 0.f, 0.f, 0.f, 0.f, 0.f, 0.f, 0.f};
  v8f o[8];
  float mrow[8], lrow[8];
#pragma unroll
  for (int i = 0; i < 8; ++i) { o[i] = zero; mrow[i] = -1e30f; lrow[i] = 0.f; }

  const float scale = 0.0721687836487032f;  // 1/sqrt(192)

  for (int j0 = 0; j0 <= m0 + 15; j0 += 32) {  // causal chunks; S%32==0 => no OOB
    const _Float16* krow0 = kh + ((size_t)h * S + j0 + nl) * D_QK;
    const _Float16* krow1 = krow0 + 16 * D_QK;
    const _Float16* vbase = vT + (size_t)(h * D_HD + nl) * S + j0 + kb16;

    // ---- scores: double-buffered K tiles across the 6 d-chunks ----
    v8f s0 = zero, s1 = zero;
    v16h b0c = load_b16(krow0 + kb16);
    v16h b1c = load_b16(krow1 + kb16);
#pragma unroll
    for (int t = 0; t < 6; ++t) {
      v16h b0n, b1n;
      if (t < 5) {                       // issue next-tile loads before current WMMAs
        b0n = load_b16(krow0 + (t + 1) * 32 + kb16);
        b1n = load_b16(krow1 + (t + 1) * 32 + kb16);
      }
      s0 = wmma_f16(qa[t], b0c, s0);
      s1 = wmma_f16(qa[t], b1c, s1);
      if (t < 5) { b0c = b0n; b1c = b1n; }
    }

    // ---- hoist first V tiles: loads fly during the softmax VALU work ----
    v16h vcur = load_b16(vbase);                     // dim block 0
    v16h vnext = load_b16(vbase + (size_t)16 * S);   // dim block 1

    // Online softmax; row M's 16 columns live in one 16-lane half => shfl_xor<16
#pragma unroll
    for (int j = 0; j < 8; ++j) {
      const int row = m0 + mb + j;
      const int c0 = j0 + nl, c1 = j0 + 16 + nl;
      const float r0 = fminf(fmaxf(s0[j], -80.f), 80.f) * scale;
      const float r1 = fminf(fmaxf(s1[j], -80.f), 80.f) * scale;
      float v0 = (c0 <= row) ? r0 : -1e30f;
      float v1 = (c1 <= row) ? r1 : -1e30f;
      float mx = fmaxf(v0, v1);
#pragma unroll
      for (int off = 1; off < 16; off <<= 1)
        mx = fmaxf(mx, __shfl_xor(mx, off, 32));
      const float mnew = fmaxf(mrow[j], mx);
      const float alpha = __expf(mrow[j] - mnew);
      const float p0 = __expf(v0 - mnew);
      const float p1 = __expf(v1 - mnew);
      float ps = p0 + p1;
#pragma unroll
      for (int off = 1; off < 16; off <<= 1)
        ps += __shfl_xor(ps, off, 32);
      lrow[j] = lrow[j] * alpha + ps;
      mrow[j] = mnew;
#pragma unroll
      for (int a = 0; a < 8; ++a) o[a][j] *= alpha;
      Pst[wv][(mb + j) * 32 + nl]      = (_Float16)p0;
      Pst[wv][(mb + j) * 32 + 16 + nl] = (_Float16)p1;
    }

    // Relayout P (C-layout f32 -> A-layout f16) through LDS (same-wave DS in-order)
    v16h pa;
    {
      const _Float16* prow = &Pst[wv][nl * 32];
#pragma unroll
      for (int i = 0; i < 8; ++i) {
        pa[i]     = prow[kb16 + i];
        pa[8 + i] = prow[kb16 + 8 + i];
      }
    }

    // ---- O += P(16x32) * V(32x128), 2-deep rotating V-tile buffer ----
#pragma unroll
    for (int ds = 0; ds < 8; ++ds) {
      v16h vuse = vcur;
      vcur = vnext;
      if (ds < 6) vnext = load_b16(vbase + (size_t)((ds + 2) * 16) * S);
      o[ds] = wmma_f16(pa, vuse, o[ds]);
    }
  }

  // Normalize and store into attn (S, H*128) row-major for the final GEMM
#pragma unroll
  for (int j = 0; j < 8; ++j) {
    const float inv = 1.f / lrow[j];
    const int r = m0 + mb + j;
#pragma unroll
    for (int ds = 0; ds < 8; ++ds) {
      attnh[(size_t)r * (H_HEADS * D_HD) + h * D_HD + ds * 16 + nl] =
          (_Float16)(o[ds][j] * inv);
    }
  }
}

// ---- Host-side pipeline ----

extern "C" void kernel_launch(void* const* d_in, const int* in_sizes, int n_in,
                              void* d_out, int out_size, void* d_ws, size_t ws_size,
                              hipStream_t stream) {
  const float* x     = (const float*)d_in[0];
  const float* W_DQ  = (const float*)d_in[1];
  const float* W_UQ  = (const float*)d_in[2];
  const float* W_QR  = (const float*)d_in[3];
  const float* W_DKV = (const float*)d_in[4];
  const float* W_UK  = (const float*)d_in[5];
  const float* W_UV  = (const float*)d_in[6];
  const float* W_KR  = (const float*)d_in[7];
  const float* W_O   = (const float*)d_in[8];
  const int S = in_sizes[0] / DMODEL;  // 2048

  char* ws = (char*)d_ws;
  size_t off = 0;
  auto alloc = [&](size_t bytes) {
    char* p = ws + off;
    off = (off + bytes + 255) & ~(size_t)255;
    return p;
  };
  _Float16* cQh   = (_Float16*)alloc((size_t)S * DLAT * 2);
  _Float16* cKVh  = (_Float16*)alloc((size_t)S * DLAT * 2);
  _Float16* kRraw = (_Float16*)alloc((size_t)S * D_HR * 2);
  _Float16* qh    = (_Float16*)alloc((size_t)H_HEADS * S * D_QK * 2);
  _Float16* kh    = (_Float16*)alloc((size_t)H_HEADS * S * D_QK * 2);
  _Float16* vT    = (_Float16*)alloc((size_t)H_HEADS * D_HD * S * 2);
  _Float16* attnh = (_Float16*)alloc((size_t)S * (H_HEADS * D_HD) * 2);
  (void)ws_size; (void)n_in; (void)out_size;

  auto blocks_for = [](int M, int N) {
    const int waves = (M >> 5) * (N >> 6);
    return (waves + 3) / 4;
  };

  // Stage 1: latents + raw rope-keys from x (A = f32, converted at load)
  gemm_wmma_kernel<float, OM_F16><<<blocks_for(S, DLAT), 128, 0, stream>>>(
      x, DMODEL, W_DQ, DMODEL, cQh, S, DLAT, DMODEL, DLAT, S);
  gemm_wmma_kernel<float, OM_F16><<<blocks_for(S, DLAT), 128, 0, stream>>>(
      x, DMODEL, W_DKV, DMODEL, cKVh, S, DLAT, DMODEL, DLAT, S);
  gemm_wmma_kernel<float, OM_F16><<<blocks_for(S, D_HR), 128, 0, stream>>>(
      x, DMODEL, W_KR, DMODEL, kRraw, S, D_HR, DMODEL, D_HR, S);

  // Stage 2: up-projections scattered straight into head-major layouts
  gemm_wmma_kernel<_Float16, OM_HEAD192_LO><<<blocks_for(S, H_HEADS * D_HD), 128, 0, stream>>>(
      cQh, DLAT, W_UQ, DLAT, qh, S, H_HEADS * D_HD, DLAT, 0, S);
  gemm_wmma_kernel<_Float16, OM_HEAD192_HI><<<blocks_for(S, H_HEADS * D_HR), 128, 0, stream>>>(
      cQh, DLAT, W_QR, DLAT, qh, S, H_HEADS * D_HR, DLAT, 0, S);
  gemm_wmma_kernel<_Float16, OM_HEAD192_LO><<<blocks_for(S, H_HEADS * D_HD), 128, 0, stream>>>(
      cKVh, DLAT, W_UK, DLAT, kh, S, H_HEADS * D_HD, DLAT, 0, S);
  gemm_wmma_kernel<_Float16, OM_VT><<<blocks_for(S, H_HEADS * D_HD), 128, 0, stream>>>(
      cKVh, DLAT, W_UV, DLAT, vT, S, H_HEADS * D_HD, DLAT, 0, S);

  // RoPE: q_R in place; k_R roped + broadcast into all heads of k
  {
    const int tq = H_HEADS * S * (D_HR / 2);
    rope_q_kernel<<<(tq + 255) / 256, 256, 0, stream>>>(qh, S);
    const int tk = S * (D_HR / 2);
    rope_kr_bcast_kernel<<<(tk + 255) / 256, 256, 0, stream>>>(kRraw, kh, S);
  }

  // Attention (flash, causal)
  {
    const int waves = H_HEADS * (S >> 4);
    mla_flash_attn_kernel<<<(waves + 3) / 4, 128, 0, stream>>>(qh, kh, vT, attnh, S);
  }

  // Output projection -> f32 d_out
  gemm_wmma_kernel<_Float16, OM_F32><<<blocks_for(S, DMODEL), 128, 0, stream>>>(
      attnh, H_HEADS * D_HD, W_O, H_HEADS * D_HD, (float*)d_out, S, DMODEL,
      H_HEADS * D_HD, DMODEL, S);
}
